// GraphQuantizer_64682207478110
// MI455X (gfx1250) — matrix operand order repaired and checked by reading, not compiled, not run on hardware
//
#include <hip/hip_runtime.h>
#include <hip/hip_fp16.h>
#include <math.h>

// ---------------------------------------------------------------------------
// Problem constants (from reference)
// ---------------------------------------------------------------------------
#define NN   16384        // nodes
#define NB   64           // graphs
#define CC   128          // feature width
#define HH   2            // heads in depth blocks
#define EHH  8            // heads in seq block
#define DEPTHN 2
#define INW  8

typedef _Float16 h16;
typedef __attribute__((ext_vector_type(16))) _Float16 v16h;
typedef __attribute__((ext_vector_type(8)))  _Float16 v8h;
typedef __attribute__((ext_vector_type(8)))  float    v8f;

// ---------------------------------------------------------------------------
// Device helpers
// ---------------------------------------------------------------------------
__device__ __forceinline__ float warp_sum(float v) {
  for (int m = 16; m > 0; m >>= 1) v += __shfl_xor(v, m, 32);
  return v;
}
__device__ __forceinline__ unsigned f2ord(float f) {
  unsigned b = __float_as_uint(f);
  return (b & 0x80000000u) ? ~b : (b | 0x80000000u);
}
__device__ __forceinline__ float ord2f(unsigned k) {
  unsigned b = (k & 0x80000000u) ? (k & 0x7fffffffu) : ~k;
  return __uint_as_float(b);
}
__device__ __forceinline__ float gelu_exact(float x) {
  return 0.5f * x * (1.0f + erff(x * 0.70710678118654752f));
}
__device__ __forceinline__ v16h cat16(v8h lo, v8h hi) {
  return __builtin_shufflevector(lo, hi, 0,1,2,3,4,5,6,7,8,9,10,11,12,13,14,15);
}
__device__ __forceinline__ void atomic_fadd(float* p, float v) {
  __hip_atomic_fetch_add(p, v, __ATOMIC_RELAXED, __HIP_MEMORY_SCOPE_AGENT);
}

// ---------------------------------------------------------------------------
// Pack f32 weight [K, Nout] row-major into WMMA B-matrix lane layout (f16).
// Block for (nt, kt): 32 lanes x 16 halves; lane l -> col nt*16+(l&15),
// k = kt*32 + (l>>4)*8 + j + (j&8), j = 0..15  (VGPR v holds halves 2v,2v+1).
// ---------------------------------------------------------------------------
__global__ void pack_b16(const float* __restrict__ W, h16* __restrict__ out,
                         int K, int Nout) {
  int tid = blockIdx.x * blockDim.x + threadIdx.x;
  int ktiles = K >> 5, ntiles = Nout >> 4;
  if (tid >= ktiles * ntiles * 32) return;
  int lane = tid & 31;
  int wid  = tid >> 5;
  int kt = wid % ktiles;
  int nt = wid / ktiles;
  int n  = (nt << 4) + (lane & 15);
  int kb = (kt << 5) + ((lane >> 4) << 3);
  h16* dst = out + (size_t)tid * 16;
#pragma unroll
  for (int j = 0; j < 16; ++j)
    dst[j] = (h16)W[(size_t)(kb + j + (j & 8)) * Nout + n];
}

// ---------------------------------------------------------------------------
// C[M,Nout] = A[M,K](f16,row-major) @ Bpacked + bias   (f32 out)
// One wave per 16x16 tile; v_wmma_f32_16x16x32_f16 over K/32 steps.
// ---------------------------------------------------------------------------
__global__ __launch_bounds__(256) void gemm_wmma(
    const h16* __restrict__ A, const h16* __restrict__ Bp,
    const float* __restrict__ bias, float* __restrict__ Cout,
    int M, int K, int Nout) {
  int wave = (blockIdx.x * blockDim.x + threadIdx.x) >> 5;
  int lane = threadIdx.x & 31;
  int mtiles = M >> 4, ntiles = Nout >> 4, ktiles = K >> 5;
  if (wave >= mtiles * ntiles) return;
  int mt = wave % mtiles, nt = wave / mtiles;
  int sel = lane >> 4, lc = lane & 15;
  const h16* arow = A + (size_t)((mt << 4) + lc) * K + sel * 8;
  const h16* bptr = Bp + (((size_t)nt * ktiles) * 32 + lane) * 16;
  v8f acc = {};
  for (int kt = 0; kt < ktiles; ++kt) {
    v8h alo = *(const v8h*)(arow);
    v8h ahi = *(const v8h*)(arow + 16);
    v8h blo = *(const v8h*)(bptr);
    v8h bhi = *(const v8h*)(bptr + 8);
    __builtin_prefetch(bptr + 512, 0, 1);
    acc = __builtin_amdgcn_wmma_f32_16x16x32_f16(
        false, cat16(alo, ahi), false, cat16(blo, bhi),
        (short)0, acc, false, false);
    arow += 32;
    bptr += 512;
  }
  int col = (nt << 4) + lc;
  float bv = bias ? bias[col] : 0.0f;
  int rbase = (mt << 4) + sel * 8;
#pragma unroll
  for (int r = 0; r < 8; ++r)
    Cout[(size_t)(rbase + r) * Nout + col] = acc[r] + bv;
}

// ---------------------------------------------------------------------------
// Fused FFN: out[rows,128] = gelu(A[rows,128] @ W1 + b1) @ W2 + b2
// W1: 128->256, W2: 256->128, both pre-packed. Per-wave 16x256 f16 LDS tile
// bridges stage1 (C-layout) -> stage2 (A-layout).
// ---------------------------------------------------------------------------
__global__ __launch_bounds__(128) void fused_ff(
    const h16* __restrict__ A, const h16* __restrict__ W1p,
    const float* __restrict__ b1, const h16* __restrict__ W2p,
    const float* __restrict__ b2, float* __restrict__ Out, int rows) {
  __shared__ __align__(16) h16 tmp[4][16 * 256];
  int wave = (blockIdx.x * blockDim.x + threadIdx.x) >> 5;
  int wiw  = threadIdx.x >> 5;
  int lane = threadIdx.x & 31;
  if (wave >= (rows >> 4)) return;
  int r0 = wave << 4;
  int sel = lane >> 4, lc = lane & 15, roff = sel * 8;

  // Preload A tile (16x128): 4 k-steps, kept in registers.
  v16h a[4];
  {
    const h16* arow = A + (size_t)(r0 + lc) * 128 + sel * 8;
#pragma unroll
    for (int kt = 0; kt < 4; ++kt) {
      v8h lo = *(const v8h*)(arow + kt * 32);
      v8h hi = *(const v8h*)(arow + kt * 32 + 16);
      a[kt] = cat16(lo, hi);
    }
  }
  h16* t = tmp[wiw];

  // Stage 1: tmp = gelu(A @ W1 + b1), 16 x 256 (f16 in LDS)
  for (int nt = 0; nt < 16; ++nt) {
    v8f acc = {};
    const h16* bptr = W1p + ((size_t)nt * 128 + lane) * 16;
#pragma unroll
    for (int kt = 0; kt < 4; ++kt) {
      v8h blo = *(const v8h*)(bptr);
      v8h bhi = *(const v8h*)(bptr + 8);
      acc = __builtin_amdgcn_wmma_f32_16x16x32_f16(
          false, a[kt], false, cat16(blo, bhi), (short)0, acc, false, false);
      bptr += 512;
    }
    int col = (nt << 4) + lc;
    float bb = b1[col];
#pragma unroll
    for (int r = 0; r < 8; ++r)
      t[(roff + r) * 256 + col] = (h16)gelu_exact(acc[r] + bb);
  }

  // All LDS stores from this wave must land before stage-2 LDS reads.
  asm volatile("s_wait_dscnt 0" ::: "memory");

  // Stage 2: Out = tmp @ W2 + b2, 16 x 128
  for (int nt = 0; nt < 8; ++nt) {
    v8f acc = {};
    const h16* bptr = W2p + ((size_t)nt * 256 + lane) * 16;
    const h16* arow = t + (size_t)lc * 256 + sel * 8;
#pragma unroll
    for (int kt = 0; kt < 8; ++kt) {
      v8h alo = *(const v8h*)(arow + kt * 32);
      v8h ahi = *(const v8h*)(arow + kt * 32 + 16);
      v8h blo = *(const v8h*)(bptr);
      v8h bhi = *(const v8h*)(bptr + 8);
      acc = __builtin_amdgcn_wmma_f32_16x16x32_f16(
          false, cat16(alo, ahi), false, cat16(blo, bhi),
          (short)0, acc, false, false);
      bptr += 512;
    }
    int col = (nt << 4) + lc;
    float bb = b2[col];
#pragma unroll
    for (int r = 0; r < 8; ++r)
      Out[(size_t)(r0 + roff + r) * 128 + col] = acc[r] + bb;
  }
}

// ---------------------------------------------------------------------------
// attn_up projection: el/er = feats[N,8] @ Ws/Wd + bias  (K=8, trivially VALU)
// ---------------------------------------------------------------------------
__global__ void up_proj(const float* __restrict__ feats,
                        const float* __restrict__ Ws, const float* __restrict__ bs,
                        const float* __restrict__ Wd, const float* __restrict__ bd,
                        float* __restrict__ el, float* __restrict__ er) {
  int i = blockIdx.x * blockDim.x + threadIdx.x;
  if (i >= NN * CC) return;
  int n = i >> 7, c = i & 127;
  const float* f = feats + (size_t)n * INW;
  float aS = bs[c], aD = bd[c];
#pragma unroll
  for (int k = 0; k < INW; ++k) {
    float fv = f[k];
    aS += fv * Ws[k * CC + c];
    aD += fv * Wd[k * CC + c];
  }
  el[i] = aS;
  er[i] = aD;
}

// ---------------------------------------------------------------------------
// GATv2 edge softmax (3 passes) -------------------------------------------
// ---------------------------------------------------------------------------
__global__ void edge_logits(const float* __restrict__ el, const float* __restrict__ er,
                            const float* __restrict__ attn, const int* __restrict__ src,
                            const int* __restrict__ dst, float* __restrict__ lgt,
                            unsigned* __restrict__ mx, int ne, int heads) {
  int wave = (blockIdx.x * blockDim.x + threadIdx.x) >> 5;
  int lane = threadIdx.x & 31;
  if (wave >= ne * heads) return;
  int e = wave / heads, h = wave % heads;
  int s = src[e], d = dst[e];
  float4 a4 = *(const float4*)(el + ((size_t)s * heads + h) * CC + lane * 4);
  float4 b4 = *(const float4*)(er + ((size_t)d * heads + h) * CC + lane * 4);
  float4 w4 = *(const float4*)(attn + (size_t)h * CC + lane * 4);
  float v = 0.0f, x;
  x = a4.x + b4.x; x = (x > 0.f) ? x : 0.2f * x; v += x * w4.x;
  x = a4.y + b4.y; x = (x > 0.f) ? x : 0.2f * x; v += x * w4.y;
  x = a4.z + b4.z; x = (x > 0.f) ? x : 0.2f * x; v += x * w4.z;
  x = a4.w + b4.w; x = (x > 0.f) ? x : 0.2f * x; v += x * w4.w;
  v = warp_sum(v);
  if (lane == 0) {
    lgt[(size_t)e * heads + h] = v;
    atomicMax(mx + (size_t)d * heads + h, f2ord(v));
  }
}

__global__ void edge_expsum(float* __restrict__ lgt, const unsigned* __restrict__ mx,
                            float* __restrict__ z, const int* __restrict__ dst,
                            int ne, int heads) {
  int i = blockIdx.x * blockDim.x + threadIdx.x;
  if (i >= ne * heads) return;
  int e = i / heads, h = i % heads;
  int d = dst[e];
  float a = expf(lgt[i] - ord2f(mx[(size_t)d * heads + h]));
  lgt[i] = a;
  atomic_fadd(z + (size_t)d * heads + h, a);
}

__global__ void edge_aggregate(const float* __restrict__ el, const float* __restrict__ lgt,
                               const float* __restrict__ z, const int* __restrict__ src,
                               const int* __restrict__ dst, float* __restrict__ out,
                               int ne, int heads) {
  int wave = (blockIdx.x * blockDim.x + threadIdx.x) >> 5;
  int lane = threadIdx.x & 31;
  if (wave >= ne * heads) return;
  int e = wave / heads, h = wave % heads;
  int s = src[e], d = dst[e];
  float w = lgt[(size_t)e * heads + h] / z[(size_t)d * heads + h];
  float4 v4 = *(const float4*)(el + ((size_t)s * heads + h) * CC + lane * 4);
  float* o = out + ((size_t)d * heads + h) * CC + lane * 4;
  atomic_fadd(o + 0, w * v4.x);
  atomic_fadd(o + 1, w * v4.y);
  atomic_fadd(o + 2, w * v4.z);
  atomic_fadd(o + 3, w * v4.w);
}

// ---------------------------------------------------------------------------
// Row LayerNorm over C=128, f32 in -> f16 out (feeds WMMA A).
// ---------------------------------------------------------------------------
__global__ void layernorm_f16(const float* __restrict__ in, const float* __restrict__ g,
                              const float* __restrict__ be, h16* __restrict__ out,
                              int rows) {
  int wave = (blockIdx.x * blockDim.x + threadIdx.x) >> 5;
  int lane = threadIdx.x & 31;
  if (wave >= rows) return;
  float4 v = *(const float4*)(in + (size_t)wave * CC + lane * 4);
  float mu = warp_sum(v.x + v.y + v.z + v.w) * (1.0f / 128.0f);
  float dx = v.x - mu, dy = v.y - mu, dz = v.z - mu, dw = v.w - mu;
  float var = warp_sum(dx * dx + dy * dy + dz * dz + dw * dw) * (1.0f / 128.0f);
  float rstd = rsqrtf(var + 1e-5f);
  float4 g4 = *(const float4*)(g + lane * 4);
  float4 b4 = *(const float4*)(be + lane * 4);
  h16* o = out + (size_t)wave * CC + lane * 4;
  o[0] = (h16)(dx * rstd * g4.x + b4.x);
  o[1] = (h16)(dy * rstd * g4.y + b4.y);
  o[2] = (h16)(dz * rstd * g4.z + b4.z);
  o[3] = (h16)(dw * rstd * g4.w + b4.w);
}

__global__ void cvt_f32_f16(const float* __restrict__ in, h16* __restrict__ out, long n) {
  long i = (long)blockIdx.x * blockDim.x + threadIdx.x;
  if (i < n) out[i] = (h16)in[i];
}

// ---------------------------------------------------------------------------
// gate[n] = y[n,:1024] . gate_W + gate_b   (one wave per node)
// ---------------------------------------------------------------------------
__global__ void gate_kernel(const float* __restrict__ y, const float* __restrict__ gW,
                            const float* __restrict__ gb, float* __restrict__ gate) {
  int wave = (blockIdx.x * blockDim.x + threadIdx.x) >> 5;
  int lane = threadIdx.x & 31;
  if (wave >= NN) return;
  const float* row = y + (size_t)wave * (EHH * CC);
  float s = 0.0f;
#pragma unroll
  for (int j = 0; j < 8; ++j) {
    float4 a = *(const float4*)(row + j * 128 + lane * 4);
    float4 w = *(const float4*)(gW + j * 128 + lane * 4);
    s += a.x * w.x + a.y * w.y + a.z * w.z + a.w * w.w;
  }
  s = warp_sum(s);
  if (lane == 0) gate[wave] = s + gb[0];
}

// ---------------------------------------------------------------------------
// Global attention pooling: graphs are 256 contiguous nodes.
// One 256-thread workgroup per graph: softmax(gate) + weighted sum + tanh.
// ---------------------------------------------------------------------------
__global__ __launch_bounds__(256) void pool_kernel(const float* __restrict__ y,
                                                   const float* __restrict__ gate,
                                                   float* __restrict__ out) {
  __shared__ float sg[256];
  __shared__ float red[256];
  int b = blockIdx.x, t = threadIdx.x;
  int n0 = b * (NN / NB);
  float g = gate[n0 + t];
  red[t] = g;
  __syncthreads();
  for (int s = 128; s > 0; s >>= 1) {
    if (t < s) red[t] = fmaxf(red[t], red[t + s]);
    __syncthreads();
  }
  float mxv = red[0];
  __syncthreads();
  float e = expf(g - mxv);
  sg[t] = e;
  red[t] = e;
  __syncthreads();
  for (int s = 128; s > 0; s >>= 1) {
    if (t < s) red[t] += red[t + s];
    __syncthreads();
  }
  float zs = red[0];
  __syncthreads();
  for (int c = t; c < EHH * CC; c += 256) {
    float acc = 0.0f;
    for (int i = 0; i < NN / NB; ++i)
      acc += sg[i] * y[(size_t)(n0 + i) * (EHH * CC) + c];
    out[(size_t)b * (EHH * CC) + c] = tanhf(acc / zs);
  }
}

// ---------------------------------------------------------------------------
// Host orchestration
// ---------------------------------------------------------------------------
static inline int cdiv(long a, long b) { return (int)((a + b - 1) / b); }

static void run_attention(const float* el, const float* er, const float* attn,
                          const int* src, const int* dst, int ne, int heads,
                          float* lgt, unsigned* mx, float* z, float* agg,
                          hipStream_t stream) {
  hipMemsetAsync(mx, 0, (size_t)NN * heads * 4, stream);
  hipMemsetAsync(z, 0, (size_t)NN * heads * 4, stream);
  hipMemsetAsync(agg, 0, (size_t)NN * heads * CC * 4, stream);
  long tw = (long)ne * heads;
  edge_logits<<<cdiv(tw * 32, 256), 256, 0, stream>>>(el, er, attn, src, dst, lgt, mx, ne, heads);
  edge_expsum<<<cdiv(tw, 256), 256, 0, stream>>>(lgt, mx, z, dst, ne, heads);
  edge_aggregate<<<cdiv(tw * 32, 256), 256, 0, stream>>>(el, lgt, z, src, dst, agg, ne, heads);
}

static void launch_gemm(const h16* A, const h16* Bp, const float* bias, float* C,
                        int M, int K, int Nout, hipStream_t stream) {
  long tiles = (long)(M >> 4) * (Nout >> 4);
  gemm_wmma<<<cdiv(tiles * 32, 256), 256, 0, stream>>>(A, Bp, bias, C, M, K, Nout);
}

extern "C" void kernel_launch(void* const* d_in, const int* in_sizes, int n_in,
                              void* d_out, int out_size, void* d_ws, size_t ws_size,
                              hipStream_t stream) {
  const float* feats    = (const float*)d_in[0];
  const int*   src      = (const int*)d_in[1];
  const int*   dst      = (const int*)d_in[2];
  const float* up_Ws    = (const float*)d_in[4];
  const float* up_bs    = (const float*)d_in[5];
  const float* up_Wd    = (const float*)d_in[6];
  const float* up_bd    = (const float*)d_in[7];
  const float* up_attn  = (const float*)d_in[8];
  const float* L_Ws     = (const float*)d_in[9];
  const float* L_bs     = (const float*)d_in[10];
  const float* L_Wd     = (const float*)d_in[11];
  const float* L_bd     = (const float*)d_in[12];
  const float* L_attn   = (const float*)d_in[13];
  const float* ff_ln_g  = (const float*)d_in[14];
  const float* ff_ln_b  = (const float*)d_in[15];
  const float* ff_W1    = (const float*)d_in[16];
  const float* ff_b1    = (const float*)d_in[17];
  const float* ff_W2    = (const float*)d_in[18];
  const float* ff_b2    = (const float*)d_in[19];
  const float* ff_W3    = (const float*)d_in[20];
  const float* ff_b3    = (const float*)d_in[21];
  const float* seq_Ws   = (const float*)d_in[22];
  const float* seq_bs   = (const float*)d_in[23];
  const float* seq_Wd   = (const float*)d_in[24];
  const float* seq_bd   = (const float*)d_in[25];
  const float* seq_attn = (const float*)d_in[26];
  const float* fs_ln_g  = (const float*)d_in[27];
  const float* fs_ln_b  = (const float*)d_in[28];
  const float* fs_W1    = (const float*)d_in[29];
  const float* fs_b1    = (const float*)d_in[30];
  const float* fs_W2    = (const float*)d_in[31];
  const float* fs_b2    = (const float*)d_in[32];
  const float* gate_W   = (const float*)d_in[33];
  const float* gate_b   = (const float*)d_in[34];
  const int ne = in_sizes[1];

  // ---- workspace bump allocator ----
  char* w = (char*)d_ws;
  auto alloc = [&](size_t bytes) -> void* {
    void* p = (void*)w;
    w += (bytes + 255) & ~(size_t)255;
    return p;
  };
  float*    el   = (float*)alloc((size_t)NN * 1024 * 4);
  float*    er   = (float*)alloc((size_t)NN * 1024 * 4);
  float*    agg  = (float*)alloc((size_t)NN * 1024 * 4);
  float*    x    = (float*)alloc((size_t)NN * CC * 4);
  h16*      xf16 = (h16*)alloc((size_t)NN * 1024 * 2);
  float*    lgt  = (float*)alloc((size_t)ne * EHH * 4);
  unsigned* mx   = (unsigned*)alloc((size_t)NN * EHH * 4);
  float*    z    = (float*)alloc((size_t)NN * EHH * 4);
  float*    gate = (float*)alloc((size_t)NN * 4);
  h16* LWs_p  = (h16*)alloc((size_t)2 * 32768 * 2);
  h16* LWd_p  = (h16*)alloc((size_t)2 * 32768 * 2);
  h16* ffW1_p = (h16*)alloc((size_t)2 * 32768 * 2);
  h16* ffW2_p = (h16*)alloc((size_t)2 * 32768 * 2);
  h16* ffW3_p = (h16*)alloc((size_t)2 * 32768 * 2);
  h16* sWs_p  = (h16*)alloc((size_t)131072 * 2);
  h16* sWd_p  = (h16*)alloc((size_t)131072 * 2);
  h16* fsW1_p = (h16*)alloc((size_t)32768 * 2);
  h16* fsW2_p = (h16*)alloc((size_t)32768 * 2);

  // ---- pack all WMMA weights (f32 -> lane-layout f16) ----
  auto pack = [&](const float* W, h16* dst, int K, int Nout) {
    int threads = (K >> 5) * (Nout >> 4) * 32;
    pack_b16<<<cdiv(threads, 256), 256, 0, stream>>>(W, dst, K, Nout);
  };
  for (int l = 0; l < DEPTHN; ++l) {
    pack(L_Ws + (size_t)l * 32768, LWs_p + (size_t)l * 32768, 128, 256);
    pack(L_Wd + (size_t)l * 32768, LWd_p + (size_t)l * 32768, 128, 256);
    pack(ff_W1 + (size_t)l * 32768, ffW1_p + (size_t)l * 32768, 128, 256);
    pack(ff_W2 + (size_t)l * 32768, ffW2_p + (size_t)l * 32768, 256, 128);
    pack(ff_W3 + (size_t)l * 32768, ffW3_p + (size_t)l * 32768, 256, 128);
  }
  pack(seq_Ws, sWs_p, 128, 1024);
  pack(seq_Wd, sWd_p, 128, 1024);
  pack(fs_W1, fsW1_p, 128, 256);
  pack(fs_W2, fsW2_p, 256, 128);

  // ---- attn_up: (one-hot @ W) projections + edge softmax, 1 head ----
  up_proj<<<cdiv((long)NN * CC, 256), 256, 0, stream>>>(feats, up_Ws, up_bs, up_Wd, up_bd, el, er);
  run_attention(el, er, up_attn, src, dst, ne, 1, lgt, mx, z, agg, stream);
  hipMemcpyAsync(x, agg, (size_t)NN * CC * 4, hipMemcpyDeviceToDevice, stream);

  // ---- depth blocks ----
  for (int l = 0; l < DEPTHN; ++l) {
    cvt_f32_f16<<<cdiv((long)NN * CC, 256), 256, 0, stream>>>(x, xf16, (long)NN * CC);
    launch_gemm(xf16, LWs_p + (size_t)l * 32768, L_bs + (size_t)l * 256, el, NN, 128, 256, stream);
    launch_gemm(xf16, LWd_p + (size_t)l * 32768, L_bd + (size_t)l * 256, er, NN, 128, 256, stream);
    run_attention(el, er, L_attn + (size_t)l * (HH * CC), src, dst, ne, HH, lgt, mx, z, agg, stream);
    // layernorm over last dim per (n,h) -> f16 A for FFN
    layernorm_f16<<<cdiv((long)NN * HH * 32, 256), 256, 0, stream>>>(
        agg, ff_ln_g + (size_t)l * CC, ff_ln_b + (size_t)l * CC, xf16, NN * HH);
    // fused gelu FFN 128->256->128 per (n,h); result is [N, H*C] row-major in el
    fused_ff<<<cdiv((long)(NN * HH / 16) * 32, 128), 128, 0, stream>>>(
        xf16, ffW1_p + (size_t)l * 32768, ff_b1 + (size_t)l * 256,
        ffW2_p + (size_t)l * 32768, ff_b2 + (size_t)l * 128, el, NN * HH);
    // x = [N, 256] @ ff_W3 + b3
    cvt_f32_f16<<<cdiv((long)NN * 256, 256), 256, 0, stream>>>(el, xf16, (long)NN * 256);
    launch_gemm(xf16, ffW3_p + (size_t)l * 32768, ff_b3 + (size_t)l * CC, x, NN, 256, 128, stream);
  }

  // ---- attn_seq (EH=8 heads) + ff_seq ----
  cvt_f32_f16<<<cdiv((long)NN * CC, 256), 256, 0, stream>>>(x, xf16, (long)NN * CC);
  launch_gemm(xf16, sWs_p, seq_bs, el, NN, 128, 1024, stream);
  launch_gemm(xf16, sWd_p, seq_bd, er, NN, 128, 1024, stream);
  run_attention(el, er, seq_attn, src, dst, ne, EHH, lgt, mx, z, agg, stream);
  layernorm_f16<<<cdiv((long)NN * EHH * 32, 256), 256, 0, stream>>>(
      agg, fs_ln_g, fs_ln_b, xf16, NN * EHH);
  fused_ff<<<cdiv((long)(NN * EHH / 16) * 32, 128), 128, 0, stream>>>(
      xf16, fsW1_p, fs_b1, fsW2_p, fs_b2, el, NN * EHH);   // el = y [N, EH*C]

  // ---- global attention pooling + tanh ----
  gate_kernel<<<cdiv((long)NN * 32, 256), 256, 0, stream>>>(el, gate_W, gate_b, gate);
  pool_kernel<<<NB, 256, 0, stream>>>(el, gate, (float*)d_out);
}
